// AdditionalLSTM_25391846654516
// MI455X (gfx1250) — matrix operand order repaired and checked by reading, not compiled, not run on hardware
//
#include <hip/hip_runtime.h>
#include <hip/hip_bf16.h>

// ---------------------------------------------------------------------------
// Bidirectional 2-layer LSTM (B=128,T=512,D=128,H=64) + dense(45) + softmax
// WMMA bf16 GEMMs + packed B-fragments + async-LDS double-buffered recurrence.
// ---------------------------------------------------------------------------

#define BB 128
#define TT 512
#define DD 128
#define HH 64
#define GG 256   // 4*H
#define CC 45
#define MTOT (BB*TT) // 65536

typedef __attribute__((ext_vector_type(16))) __bf16 v16bf;
typedef __attribute__((ext_vector_type(8)))  __bf16 v8bf;
typedef __attribute__((ext_vector_type(8)))  float  v8f;
typedef __attribute__((ext_vector_type(4)))  int    v4i;

union V16U { v16bf v; v8bf h[2]; };

#if defined(__has_builtin)
#if __has_builtin(__builtin_amdgcn_global_load_async_to_lds_b128)
#define USE_ASYNC 1
#endif
#endif

__device__ __forceinline__ void wait_async0() {
#if defined(__has_builtin) && __has_builtin(__builtin_amdgcn_s_wait_asynccnt)
    __builtin_amdgcn_s_wait_asynccnt(0);
#else
    asm volatile("s_wait_asynccnt 0" ::: "memory");
#endif
}

// --- A-fragment (16x32 bf16) from global, row-major, leading dim `ld`.
// Lane layout (ISA 7.12.2): M = lane%16 ; elems 0..7 = K kbase+k0+0..7,
// elems 8..15 = K kbase+k0+16..23, where k0 = (lane>=16)?8:0.
__device__ __forceinline__ v16bf load_a_g(const __bf16* __restrict__ A, int row,
                                          int ld, int kbase, int lane) {
    int k0 = kbase + ((lane & 16) ? 8 : 0);
    const __bf16* p = A + (long)row * ld + k0;
    V16U u;
    u.h[0] = *(const v8bf*)(p);
    u.h[1] = *(const v8bf*)(p + 16);
    return u.v;
}

// --- A-fragment from LDS h-state (16x64 bf16, row-major ld=64).
__device__ __forceinline__ v16bf load_a_lds(const __bf16* h, int kbase, int lane) {
    int m  = lane & 15;
    int k0 = kbase + ((lane & 16) ? 8 : 0);
    const __bf16* p = h + m * HH + k0;
    V16U u;
    u.h[0] = *(const v8bf*)(p);
    u.h[1] = *(const v8bf*)(p + 16);
    return u.v;
}

// ---------------------------------------------------------------------------
// f32 -> bf16 conversion
// ---------------------------------------------------------------------------
__global__ void __launch_bounds__(256) k_f32_to_bf16(const float* __restrict__ in,
                                                     __bf16* __restrict__ out, long n) {
    long i = (long)blockIdx.x * 256 + threadIdx.x;
    if (i < n) out[i] = (__bf16)in[i];
}

// ---------------------------------------------------------------------------
// Pack a (Kc*32 x 256) f32 weight matrix into bf16 WMMA B-fragment order:
//   Wp[((kc*16 + ntile)*32 + lane)*16 + e] =
//       W[(kc*32 + kb + (e<8 ? e : e+8))*256 + ntile*16 + lane%16],  kb = (lane&16)?8:0
// so a B-fragment is one contiguous 32B load per lane.
// ---------------------------------------------------------------------------
__global__ void __launch_bounds__(256) k_pack_b(const float* __restrict__ W,
                                                __bf16* __restrict__ Wp, int total) {
    int idx = blockIdx.x * 256 + threadIdx.x;
    if (idx >= total) return;
    int e     = idx & 15;
    int lane  = (idx >> 4) & 31;
    int frag  = idx >> 9;
    int ntile = frag & 15;
    int kc    = frag >> 4;
    int kb    = (lane & 16) ? 8 : 0;
    int n     = ntile * 16 + (lane & 15);
    Wp[idx] = (__bf16)W[(kc * 32 + kb + (e < 8 ? e : e + 8)) * GG + n];
}

// ---------------------------------------------------------------------------
// xz = A(bf16, 65536 x 128) @ W(packed bf16, 128 x 256) + bias.
// Each wave: one 16-row M-tile x 4 N-tiles; K=128 -> 4 chained WMMAs/tile,
// A fragment reused across the 4 N-tiles.
// ---------------------------------------------------------------------------
__global__ void __launch_bounds__(256) k_gemm_xz(const __bf16* __restrict__ A,
                                                 const __bf16* __restrict__ Wpf,
                                                 const __bf16* __restrict__ Wpb,
                                                 const float* __restrict__ biasf,
                                                 const float* __restrict__ biasb,
                                                 float* __restrict__ outf,
                                                 float* __restrict__ outb) {
    const int lane = threadIdx.x & 31;
    const int wave = threadIdx.x >> 5;
    const v16bf* Wp   = (const v16bf*)(blockIdx.y ? Wpb : Wpf);
    const float* bias = blockIdx.y ? biasb : biasf;
    float* out        = blockIdx.y ? outb : outf;

    int id    = blockIdx.x * 8 + wave;   // 16384 wave-tiles
    int ntg   = id & 3;                  // group of 4 N-tiles
    int mtile = id >> 2;                 // 4096 M-tiles
    int row   = mtile * 16 + (lane & 15);

    v8f acc[4] = {};
#pragma unroll
    for (int kc = 0; kc < 4; ++kc) {
        v16bf a = load_a_g(A, row, DD, kc * 32, lane);
#pragma unroll
        for (int j = 0; j < 4; ++j) {
            v16bf b = Wp[(kc * 16 + ntg * 4 + j) * 32 + lane];
            acc[j] = __builtin_amdgcn_wmma_f32_16x16x32_bf16(false, a, false, b,
                                                             (short)0, acc[j], false, false);
        }
    }
    int mrow0 = mtile * 16 + ((lane & 16) ? 8 : 0);
#pragma unroll
    for (int j = 0; j < 4; ++j) {
        int n    = (ntg * 4 + j) * 16 + (lane & 15);
        float bv = bias[n];
#pragma unroll
        for (int r = 0; r < 8; ++r)
            out[(long)(mrow0 + r) * GG + n] = acc[j][r] + bv;
    }
}

// ---------------------------------------------------------------------------
// Recurrence for one layer, both directions.
// blockIdx.x: [0..7] = fwd batch tiles, [8..15] = bwd batch tiles.
// h (bf16) / c (f32) live in LDS; packed U B-fragments preloaded to VGPRs.
// xz slice (16x256 f32) double-buffered in LDS via async global->LDS loads
// issued one timestep ahead.  Per step per wave: 4 v_wmma_f32_16x16x32_bf16.
// ---------------------------------------------------------------------------
__global__ void __launch_bounds__(256) k_lstm_rec(const float* __restrict__ xzf,
                                                  const float* __restrict__ xzb,
                                                  const __bf16* __restrict__ Upf,
                                                  const __bf16* __restrict__ Upb,
                                                  __bf16* __restrict__ hout) {
    __shared__ __bf16 h_lds[16 * HH];        // 2 KB
    __shared__ float  c_lds[16 * HH];        // 4 KB
    __shared__ float  zact[16 * GG];         // 16 KB (activated gates)
#if USE_ASYNC
    __shared__ float  xzbuf[2][16 * GG];     // 2 x 16 KB double buffer
#endif

    const int tid   = threadIdx.x;
    const int lane  = tid & 31;
    const int wave  = tid >> 5;
    const int tileb = blockIdx.x & 7;
    const int dir   = blockIdx.x >> 3;
    const int b0    = tileb * 16;
    const float*  xz = dir ? xzb : xzf;
    const v16bf*  Up = (const v16bf*)(dir ? Upb : Upf);

    for (int e = tid; e < 16 * HH; e += 256) {
        c_lds[e] = 0.f;
        h_lds[e] = (__bf16)0.f;
    }

    // Preload this wave's packed U fragments (N cols [wave*32,wave*32+32), K=64).
    v16bf ub[2][2];
#pragma unroll
    for (int nt = 0; nt < 2; ++nt)
#pragma unroll
        for (int kc = 0; kc < 2; ++kc)
            ub[nt][kc] = Up[(kc * 16 + wave * 2 + nt) * 32 + lane];

    const int mrow   = (lane & 16) ? 8 : 0;
    const int g_wave = wave >> 1;            // 0=i 1=f 2=g 3=o

#if USE_ASYNC
    // Prefetch xz tile for t=0: 1024 chunks of 16B, 4 per thread.
    {
        int tt0 = dir ? (TT - 1) : 0;
#pragma unroll
        for (int i = 0; i < 4; ++i) {
            int l = tid + 256 * i;
            int m = l >> 6;
            int o = (l & 63) * 4;
            const float* g = xz + ((long)(b0 + m) * TT + tt0) * GG + o;
            float* s = &xzbuf[0][m * GG + o];
            __builtin_amdgcn_global_load_async_to_lds_b128((v4i*)g, (v4i*)s, 0, 0);
        }
    }
#endif
    int cur = 0;

    for (int t = 0; t < TT; ++t) {
        int tt = dir ? (TT - 1 - t) : t;

#if USE_ASYNC
        wait_async0();                 // own async loads done
#endif
        __syncthreads();               // (a) tile-for-t visible; h_lds settled

#if USE_ASYNC
        if (t + 1 < TT) {              // prefetch t+1 into the other buffer
            int tn = dir ? (TT - 2 - t) : (t + 1);
#pragma unroll
            for (int i = 0; i < 4; ++i) {
                int l = tid + 256 * i;
                int m = l >> 6;
                int o = (l & 63) * 4;
                const float* g = xz + ((long)(b0 + m) * TT + tn) * GG + o;
                float* s = &xzbuf[cur ^ 1][m * GG + o];
                __builtin_amdgcn_global_load_async_to_lds_b128((v4i*)g, (v4i*)s, 0, 0);
            }
        }
#endif

        v16bf a0 = load_a_lds(h_lds, 0, lane);
        v16bf a1 = load_a_lds(h_lds, 32, lane);

#pragma unroll
        for (int nt = 0; nt < 2; ++nt) {
            int n = wave * 32 + nt * 16 + (lane & 15);
            v8f acc;
#pragma unroll
            for (int r = 0; r < 8; ++r)
#if USE_ASYNC
                acc[r] = xzbuf[cur][(mrow + r) * GG + n];
#else
                acc[r] = xz[((long)(b0 + mrow + r) * TT + tt) * GG + n];
#endif
            acc = __builtin_amdgcn_wmma_f32_16x16x32_bf16(false, a0, false, ub[nt][0],
                                                          (short)0, acc, false, false);
            acc = __builtin_amdgcn_wmma_f32_16x16x32_bf16(false, a1, false, ub[nt][1],
                                                          (short)0, acc, false, false);
#pragma unroll
            for (int r = 0; r < 8; ++r) {
                float z   = acc[r];
                float act = (g_wave == 2) ? fmaxf(z, 0.f)
                                          : 1.f / (1.f + __expf(-z));
                zact[(mrow + r) * GG + n] = act;
            }
        }
        __syncthreads();               // (b) gates ready

        // State update: c = f*c + i*relu(g) ; h = o*relu(c)
#pragma unroll
        for (int r = 0; r < 4; ++r) {
            int e = tid + 256 * r;     // 1024 state elems
            int m = e >> 6, j = e & 63;
            float zi = zact[m * GG + j];
            float zf = zact[m * GG + 64 + j];
            float zg = zact[m * GG + 128 + j];
            float zo = zact[m * GG + 192 + j];
            float cc = zf * c_lds[e] + zi * zg;
            c_lds[e] = cc;
            float hh = zo * fmaxf(cc, 0.f);
            h_lds[e] = (__bf16)hh;
            hout[((long)(b0 + m) * TT + tt) * (2 * HH) + dir * HH + j] = (__bf16)hh;
        }
        cur ^= 1;
        // next iteration's barrier (a) orders h_lds writes vs. A-frag reads
    }
}

// ---------------------------------------------------------------------------
// logits = h2 @ Wd + bd ; softmax.  One row (b,t) per thread; Wd in LDS.
// ---------------------------------------------------------------------------
__global__ void __launch_bounds__(256) k_dense_softmax(const __bf16* __restrict__ h2,
                                                       const float* __restrict__ Wd,
                                                       const float* __restrict__ bd,
                                                       float* __restrict__ out) {
    __shared__ float wlds[2 * HH * CC + CC];
    for (int e = threadIdx.x; e < 2 * HH * CC; e += 256) wlds[e] = Wd[e];
    if (threadIdx.x < CC) wlds[2 * HH * CC + threadIdx.x] = bd[threadIdx.x];
    __syncthreads();

    long row = (long)blockIdx.x * 256 + threadIdx.x;   // MTOT rows
    float acc[CC];
#pragma unroll
    for (int c = 0; c < CC; ++c) acc[c] = wlds[2 * HH * CC + c];

    const __bf16* hp = h2 + row * (2 * HH);
    for (int k = 0; k < 2 * HH; ++k) {
        float hv = (float)hp[k];
#pragma unroll
        for (int c = 0; c < CC; ++c) acc[c] += hv * wlds[k * CC + c];
    }
    float mx = acc[0];
#pragma unroll
    for (int c = 1; c < CC; ++c) mx = fmaxf(mx, acc[c]);
    float s = 0.f;
#pragma unroll
    for (int c = 0; c < CC; ++c) { acc[c] = __expf(acc[c] - mx); s += acc[c]; }
    float inv = 1.f / s;
#pragma unroll
    for (int c = 0; c < CC; ++c) out[row * CC + c] = acc[c] * inv;
}

// ---------------------------------------------------------------------------
extern "C" void kernel_launch(void* const* d_in, const int* in_sizes, int n_in,
                              void* d_out, int out_size, void* d_ws, size_t ws_size,
                              hipStream_t stream) {
    const float* x   = (const float*)d_in[0];
    const float* W1f = (const float*)d_in[1];
    const float* U1f = (const float*)d_in[2];
    const float* b1f = (const float*)d_in[3];
    const float* W1b = (const float*)d_in[4];
    const float* U1b = (const float*)d_in[5];
    const float* b1b = (const float*)d_in[6];
    const float* W2f = (const float*)d_in[7];
    const float* U2f = (const float*)d_in[8];
    const float* b2f = (const float*)d_in[9];
    const float* W2b = (const float*)d_in[10];
    const float* U2b = (const float*)d_in[11];
    const float* b2b = (const float*)d_in[12];
    const float* Wd  = (const float*)d_in[13];
    const float* bd  = (const float*)d_in[14];
    float* out = (float*)d_out;

    // Workspace carve-out (bytes)
    char* ws = (char*)d_ws;
    float*  xz_f  = (float*)(ws);                     //  67,108,864
    float*  xz_b  = (float*)(ws + 67108864);          //  67,108,864
    __bf16* x_bf  = (__bf16*)(ws + 134217728);        //  16,777,216
    __bf16* wbf   = (__bf16*)(ws + 150994944);        //  packed weights
    __bf16* W1f_p = wbf + 0;                          //  32768 elems each
    __bf16* W1b_p = wbf + 32768;
    __bf16* W2f_p = wbf + 65536;
    __bf16* W2b_p = wbf + 98304;
    __bf16* U1f_p = wbf + 131072;                     //  16384 elems each
    __bf16* U1b_p = wbf + 147456;
    __bf16* U2f_p = wbf + 163840;
    __bf16* U2b_p = wbf + 180224;                     //  ends at +393216 B
    __bf16* h1_bf = (__bf16*)(ws + 151388160);        //  16,777,216
    __bf16* h2_bf = (__bf16*)(ws + 168165376);        //  16,777,216

    dim3 blk(256);

    // Convert x once; pack all weights (f32 -> bf16 fragment order) once.
    k_f32_to_bf16<<<(MTOT * DD + 255) / 256, blk, 0, stream>>>(x, x_bf, (long)MTOT * DD);
    k_pack_b<<<128, blk, 0, stream>>>(W1f, W1f_p, DD * GG);
    k_pack_b<<<128, blk, 0, stream>>>(W1b, W1b_p, DD * GG);
    k_pack_b<<<128, blk, 0, stream>>>(W2f, W2f_p, 2 * HH * GG);
    k_pack_b<<<128, blk, 0, stream>>>(W2b, W2b_p, 2 * HH * GG);
    k_pack_b<<<64,  blk, 0, stream>>>(U1f, U1f_p, HH * GG);
    k_pack_b<<<64,  blk, 0, stream>>>(U1b, U1b_p, HH * GG);
    k_pack_b<<<64,  blk, 0, stream>>>(U2f, U2f_p, HH * GG);
    k_pack_b<<<64,  blk, 0, stream>>>(U2b, U2b_p, HH * GG);

    // Layer 1: input projection (both dirs) then recurrence (both dirs).
    k_gemm_xz<<<dim3(2048, 2), blk, 0, stream>>>(x_bf, W1f_p, W1b_p, b1f, b1b, xz_f, xz_b);
    k_lstm_rec<<<16, blk, 0, stream>>>(xz_f, xz_b, U1f_p, U1b_p, h1_bf);

    // Layer 2 on h1 (concat fwd|bwd, 2H=128 = same K as layer 1).
    k_gemm_xz<<<dim3(2048, 2), blk, 0, stream>>>(h1_bf, W2f_p, W2b_p, b2f, b2b, xz_f, xz_b);
    k_lstm_rec<<<16, blk, 0, stream>>>(xz_f, xz_b, U2f_p, U2b_p, h2_bf);

    // Dense + softmax.
    k_dense_softmax<<<MTOT / 256, blk, 0, stream>>>(h2_bf, Wd, bd, out);
}
